// CustomRelation_15530601742884
// MI455X (gfx1250) — compile-verified, matrix operand order checked by reading, not run on hardware
//
#include <hip/hip_runtime.h>
#include <hip/hip_bf16.h>

#define Bc 16
#define Sc 2048
#define Hc 1024
#define Dc 64
#define Rc 2048
#define Nc 128               // 2*D output columns
#define Mc (Bc * Sc)         // 32768 rows

// W staged in LDS as WMMA-ready 1KB fragments: frag f = kb*8 + t  (kb = K/32, t = N/16)
// lane's 16 bf16: elems 0..7 at f*1024 + lane*16, elems 8..15 at f*1024 + 512 + lane*16
#define LDS_BYTES ((Hc / 32) * 8 * 1024)   // 256 KB of 320KB WGP LDS

typedef __attribute__((ext_vector_type(16))) __bf16 v16bf;
typedef __attribute__((ext_vector_type(8)))  __bf16 v8bf;
typedef __attribute__((ext_vector_type(8)))  float  v8f;

// Load one 16-bit A-fragment (ISA 16-bit A 16x32 layout):
// elems 0..7 -> K = hi*8 + j ; elems 8..15 -> K = 16 + hi*8 + j
__device__ __forceinline__ v16bf load_a_frag(const float* __restrict__ arow, int hi) {
    v8f lo = *reinterpret_cast<const v8f*>(arow + hi * 8);
    v8f hv = *reinterpret_cast<const v8f*>(arow + 16 + hi * 8);
    v16bf a;
#pragma unroll
    for (int j = 0; j < 8; ++j) {
        a[j]     = (__bf16)lo[j];    // native v_cvt_pk_bf16_f32 (RNE)
        a[8 + j] = (__bf16)hv[j];
    }
    return a;
}

// ---------------------------------------------------------------------------
// Fused GEMM (bf16 WMMA, f32 acc) + bias + RoPE -> qw, kw (f32).
// W staged once per block into LDS in fragment order (immediate-offset ds_loads,
// no VALU between WMMAs -> no WMMA->VALU hazard NOPs).
// Each wave: 32(M) x 128(N) tile = two 16-row subtiles sharing B fragments.
// ---------------------------------------------------------------------------
__global__ __launch_bounds__(256) void gemm_rope(const float* __restrict__ hidden,
                                                 const float* __restrict__ W,
                                                 const float* __restrict__ bias,
                                                 const int* __restrict__ pos_ids,
                                                 float* __restrict__ qw,
                                                 float* __restrict__ kw) {
    extern __shared__ __bf16 smem[];
    char* smemc = reinterpret_cast<char*>(smem);

    // ---- Stage W -> LDS in fragment order.
    // Chunk c covers column n, K = kc*8 .. kc*8+7 (16 contiguous LDS bytes).
    // Consecutive tids share kc and walk n -> global reads fully coalesced.
    for (int c = threadIdx.x; c < (Hc / 8) * Nc; c += 256) {
        const int n  = c & (Nc - 1);
        const int kc = c >> 7;
        const int k0 = kc * 8;
        v8bf h;
#pragma unroll
        for (int j = 0; j < 8; ++j)
            h[j] = (__bf16)W[(size_t)(k0 + j) * Nc + n];
        const int kb     = kc >> 2;        // K-block of 32
        const int hi     = kc & 1;
        const int half   = (kc >> 1) & 1;
        const int t      = n >> 4;
        const int lane15 = n & 15;
        const int addr = ((kb * 8 + t) << 10) + (half << 9) + ((hi * 16 + lane15) << 4);
        *reinterpret_cast<v8bf*>(smemc + addr) = h;
    }
    __syncthreads();

    const int lane   = threadIdx.x & 31;
    const int wave   = threadIdx.x >> 5;
    const int lane15 = lane & 15;
    const int hi     = lane >> 4;

    const int m0 = (blockIdx.x * 8 + wave) * 32;     // wave's 32-row tile

    const float* arow0 = hidden + (size_t)(m0 + lane15) * Hc;
    const float* arow1 = hidden + (size_t)(m0 + 16 + lane15) * Hc;

    v8f acc[2][8];
#pragma unroll
    for (int u = 0; u < 2; ++u)
#pragma unroll
        for (int t = 0; t < 8; ++t)
            acc[u][t] = (v8f){0.f,0.f,0.f,0.f,0.f,0.f,0.f,0.f};

    const char* lbase = smemc + lane * 16;           // single DS address register
    for (int kb = 0; kb < Hc / 32; ++kb) {
        v16bf a0 = load_a_frag(arow0 + kb * 32, hi);
        v16bf a1 = load_a_frag(arow1 + kb * 32, hi);
        const char* fb = lbase + kb * 8192;

#pragma unroll
        for (int t = 0; t < 8; ++t) {
            // immediate offsets: t*1024 (+512) <= 0x1E00, fits DS OFFSET field
            v8bf blo = *reinterpret_cast<const v8bf*>(fb + t * 1024);
            v8bf bhi = *reinterpret_cast<const v8bf*>(fb + t * 1024 + 512);
            v16bf bfrag = __builtin_shufflevector(blo, bhi,
                0, 1, 2, 3, 4, 5, 6, 7, 8, 9, 10, 11, 12, 13, 14, 15);
            acc[0][t] = __builtin_amdgcn_wmma_f32_16x16x32_bf16(
                false, a0, false, bfrag, (short)0, acc[0][t], false, false);
            acc[1][t] = __builtin_amdgcn_wmma_f32_16x16x32_bf16(
                false, a1, false, bfrag, (short)0, acc[1][t], false, false);
        }
    }

    // ---- Epilogue: bias + RoPE + scatter to qw/kw ---------------------------
    // C/D layout: N = lane15; VGPR r holds M = (tile) + r + hi*8.
#pragma unroll
    for (int t = 0; t < 8; ++t) {
        const int n = t * 16 + lane15;               // 0..127
        const int c = n & (Dc - 1);                  // column within q or k
        // inv_freq = 10000^(-2*(c/2)/D) = exp(-(c>>1)/32 * ln(10000))
        const float inv_freq = __expf(-(float)(c >> 1) * (9.210340371976184f / 32.0f));
        const float bval = bias[n];
#pragma unroll
        for (int u = 0; u < 2; ++u) {
#pragma unroll
            for (int r = 0; r < 8; ++r) {
                const int m = m0 + u * 16 + r + hi * 8;
                float x = acc[u][t][r] + bval;
                // partner column n^1 lives in lane^1 (same VGPR, same t)
                float p = __shfl_xor(x, 1, 32);
                float x2 = (n & 1) ? p : -p;         // x2[2j]=-x[2j+1], x2[2j+1]=x[2j]
                float ang = (float)pos_ids[m] * inv_freq;
                float sv = __sinf(ang);
                float cv = __cosf(ang);
                float out = x * cv + x2 * sv;
                if (n < Dc) qw[(size_t)m * Dc + c] = out;
                else        kw[(size_t)m * Dc + c] = out;
            }
        }
    }
}

// ---------------------------------------------------------------------------
// Gather 4 rows per relation, 128-dim dot, mask, /sqrt(D).
// One wave per relation; lane covers dims (2l, 2l+1) of each 64-dim row.
// ---------------------------------------------------------------------------
__global__ __launch_bounds__(256) void gather_dot(const float* __restrict__ qw,
                                                  const float* __restrict__ kw,
                                                  const int* __restrict__ rel,
                                                  const float* __restrict__ mask,
                                                  float* __restrict__ out) {
    const int lane = threadIdx.x & 31;
    const int wave = threadIdx.x >> 5;
    const int g = blockIdx.x * 8 + wave;             // relation id in [0, B*R)
    if (g >= Bc * Rc) return;
    const int b = g / Rc;

    const int i0 = rel[g * 4 + 0];
    const int i1 = rel[g * 4 + 1];
    const int i2 = rel[g * 4 + 2];
    const int i3 = rel[g * 4 + 3];

    const float2* q0 = reinterpret_cast<const float2*>(qw + ((size_t)(b * Sc + i0)) * Dc) + lane;
    const float2* q1 = reinterpret_cast<const float2*>(qw + ((size_t)(b * Sc + i1)) * Dc) + lane;
    const float2* k0 = reinterpret_cast<const float2*>(kw + ((size_t)(b * Sc + i2)) * Dc) + lane;
    const float2* k1 = reinterpret_cast<const float2*>(kw + ((size_t)(b * Sc + i3)) * Dc) + lane;

    float2 a0 = *q0, a1 = *q1, c0 = *k0, c1 = *k1;
    float sum = a0.x * c0.x + a0.y * c0.y + a1.x * c1.x + a1.y * c1.y;

#pragma unroll
    for (int off = 16; off >= 1; off >>= 1)
        sum += __shfl_xor(sum, off, 32);

    if (lane == 0) {
        float lm = mask[g];
        out[g] = (sum + (1.0f - lm) * -1.0e12f) * 0.125f;  // 1/sqrt(64)
    }
}

// ---------------------------------------------------------------------------
extern "C" void kernel_launch(void* const* d_in, const int* in_sizes, int n_in,
                              void* d_out, int out_size, void* d_ws, size_t ws_size,
                              hipStream_t stream) {
    const float* hidden  = (const float*)d_in[0];   // (B,S,H) f32
    const float* W       = (const float*)d_in[1];   // (H,2D)  f32
    const float* bias    = (const float*)d_in[2];   // (2D,)   f32
    const int*   pos_ids = (const int*)d_in[3];     // (B,S)   i32
    const int*   rel     = (const int*)d_in[4];     // (B,R,4) i32
    const float* mask    = (const float*)d_in[5];   // (B,R)   f32
    float* out = (float*)d_out;                     // (B,R)   f32

    char* ws = (char*)d_ws;
    float* qw = (float*)ws;                         // 32768*64*4 = 8 MB
    float* kw = qw + (size_t)Mc * Dc;               // 8 MB

    // allow 256KB dynamic LDS (CDNA5: up to 320KB per workgroup)
    (void)hipFuncSetAttribute(reinterpret_cast<const void*>(gemm_rope),
                              hipFuncAttributeMaxDynamicSharedMemorySize, LDS_BYTES);

    // 1) fused GEMM + bias + RoPE (8 waves/block, 256 M-rows per block)
    gemm_rope<<<Mc / 256, 256, LDS_BYTES, stream>>>(hidden, W, bias, pos_ids, qw, kw);

    // 2) gather + dot + mask (8 relations per 256-thread block)
    gather_dot<<<(Bc * Rc) / 8, 256, 0, stream>>>(qw, kw, rel, mask, out);
}
